// FixedNMP_18992345383582
// MI455X (gfx1250) — matrix operand order repaired and checked by reading, not compiled, not run on hardware
//
#include <hip/hip_runtime.h>
#include <hip/hip_bf16.h>

// Problem constants (match reference)
#define BATCH 128
#define NNODE 512
#define DDIM  64
#define NITER 3
#define MTILES 32              // NNODE / 16
#define TILES_TOTAL (BATCH * MTILES)   // 4096 (b, mt) tiles

typedef __attribute__((ext_vector_type(16))) __bf16 v16bf;
typedef __attribute__((ext_vector_type(8)))  float  v8f;

// ---------------------------------------------------------------------------
// CDNA5 WMMA: D = A(16x32 bf16) x B(32x16 bf16) + C(16x16 f32)
// ---------------------------------------------------------------------------
__device__ __forceinline__ v8f wmma_bf16(v16bf a, v16bf b, v8f c) {
  return __builtin_amdgcn_wmma_f32_16x16x32_bf16(
      /*neg_a=*/false, a, /*neg_b=*/false, b,
      /*c_mod=*/(short)0, c, /*reuse_a=*/false, /*reuse_b=*/false);
}

// Pre-packed B fragment: frag id = nt*KT + kt, 32 lanes x 16 bf16 each.
__device__ __forceinline__ v16bf load_b_frag(const __bf16* __restrict__ w,
                                             int KT, int kt, int nt, int lane) {
  return *(const v16bf*)(w + (((size_t)(nt * KT + kt)) * 32 + lane) * 16);
}

// Build an A fragment from a row of f32 h. A layout (ISA 7.12.2, 16-bit A 16x32):
// lane L: M = L&15, half = L>>4; e<8 -> K = kt*32 + 8*half + e
//                               e>=8 -> K = kt*32 + 16 + 8*half + (e-8)
__device__ __forceinline__ v16bf make_a_frag(const float* __restrict__ hrow,
                                             int kt, int half) {
  const float4* q0 = (const float4*)(hrow + kt * 32 + half * 8);
  const float4* q1 = (const float4*)(hrow + kt * 32 + 16 + half * 8);
  float4 u0 = q0[0], u1 = q0[1], u2 = q1[0], u3 = q1[1];
  v16bf f;
  f[0]  = (__bf16)u0.x; f[1]  = (__bf16)u0.y; f[2]  = (__bf16)u0.z; f[3]  = (__bf16)u0.w;
  f[4]  = (__bf16)u1.x; f[5]  = (__bf16)u1.y; f[6]  = (__bf16)u1.z; f[7]  = (__bf16)u1.w;
  f[8]  = (__bf16)u2.x; f[9]  = (__bf16)u2.y; f[10] = (__bf16)u2.z; f[11] = (__bf16)u2.w;
  f[12] = (__bf16)u3.x; f[13] = (__bf16)u3.y; f[14] = (__bf16)u3.z; f[15] = (__bf16)u3.w;
  return f;
}

// ---------------------------------------------------------------------------
// Weight prep: convert f32 weights to bf16, packed per-lane in B-fragment
// order. B layout (same K order as A): lane L holds
// N = L&15; e<8 -> K = kt*32 + 8*(L>>4) + e; e>=8 -> K = kt*32 + 16 + 8*(L>>4) + (e-8).
// transpose=1 packs src^T (for w_ih / w_hh which the reference multiplies
// as inp @ W.T). src_ld is the row length of src in memory.
// ---------------------------------------------------------------------------
__global__ void nmp_pack_b(const float* __restrict__ src, __bf16* __restrict__ dst,
                           int KT, int src_ld, int N_total, int transpose) {
  int frag = blockIdx.x;           // NT*KT frags
  int nt = frag / KT, kt = frag % KT;
  int lane = threadIdx.x;          // 32 threads
  int half = lane >> 4;
  int n = nt * 16 + (lane & 15);
#pragma unroll
  for (int e = 0; e < 16; ++e) {
    int k = kt * 32 + (e < 8 ? half * 8 + e : 16 + half * 8 + (e - 8));
    float v = transpose ? src[(size_t)n * src_ld + k]
                        : src[(size_t)k * src_ld + n];
    dst[((size_t)frag * 32 + lane) * 16 + e] = (__bf16)v;
  }
  (void)N_total;
}

// ---------------------------------------------------------------------------
// Embedding: h = relu(x @ emb_w + emb_b), K=7 -> plain VALU.
// thread t = bn*64 + d  (coalesced stores, x broadcast within wave)
// ---------------------------------------------------------------------------
__global__ void nmp_embed(const float* __restrict__ x, const float* __restrict__ w,
                          const float* __restrict__ bias, float* __restrict__ h) {
  int t = blockIdx.x * blockDim.x + threadIdx.x;
  if (t >= BATCH * NNODE * DDIM) return;
  int d = t & 63;
  int bn = t >> 6;
  const float* xp = x + (size_t)bn * 7;
  float acc = bias[d];
#pragma unroll
  for (int f = 0; f < 7; ++f) acc = fmaf(xp[f], w[f * DDIM + d], acc);
  h[t] = fmaxf(acc, 0.f);
}

// ---------------------------------------------------------------------------
// Residual FC block: hout = hin + relu(hin @ W + bias).  4 waves/block,
// one wave per (b, mt) tile; 8 WMMA per wave.
// ---------------------------------------------------------------------------
__global__ __launch_bounds__(128) void nmp_resgemm(
    const float* __restrict__ hin, const __bf16* __restrict__ wpack,
    const float* __restrict__ bias, float* __restrict__ hout) {
  int wave = threadIdx.x >> 5, lane = threadIdx.x & 31;
  int tile = blockIdx.x * 4 + wave;
  int b = tile >> 5, mt = tile & 31;
  int half = lane >> 4, l15 = lane & 15;
  const float* hrow = hin + ((size_t)(b * NNODE + mt * 16 + l15)) * DDIM;
  v16bf a0 = make_a_frag(hrow, 0, half);
  v16bf a1 = make_a_frag(hrow, 1, half);
  int row0 = mt * 16 + 8 * half;
#pragma unroll
  for (int nt = 0; nt < 4; ++nt) {
    v8f acc = {};
    acc = wmma_bf16(a0, load_b_frag(wpack, 2, 0, nt, lane), acc);
    acc = wmma_bf16(a1, load_b_frag(wpack, 2, 1, nt, lane), acc);
    int c = nt * 16 + l15;
    float bi = bias[c];
#pragma unroll
    for (int r = 0; r < 8; ++r) {
      size_t idx = ((size_t)(b * NNODE + row0 + r)) * DDIM + c;
      hout[idx] = hin[idx] + fmaxf(acc[r] + bi, 0.f);
    }
  }
}

// ---------------------------------------------------------------------------
// Message MLP + masked column reduction:
//   m = h + relu(h @ Wm + bm);  partial[b,mt,half,c] = sum_r mask[row]*m[row,c]
// m is never materialized.
// ---------------------------------------------------------------------------
__global__ __launch_bounds__(128) void nmp_msg(
    const float* __restrict__ hin, const __bf16* __restrict__ wpack,
    const float* __restrict__ bias, const float* __restrict__ mask,
    float* __restrict__ pbuf) {
  int wave = threadIdx.x >> 5, lane = threadIdx.x & 31;
  int tile = blockIdx.x * 4 + wave;
  int b = tile >> 5, mt = tile & 31;
  int half = lane >> 4, l15 = lane & 15;
  const float* hrow = hin + ((size_t)(b * NNODE + mt * 16 + l15)) * DDIM;
  v16bf a0 = make_a_frag(hrow, 0, half);
  v16bf a1 = make_a_frag(hrow, 1, half);
  int row0 = mt * 16 + 8 * half;
  float mk[8];
#pragma unroll
  for (int r = 0; r < 8; ++r) mk[r] = mask[b * NNODE + row0 + r];
#pragma unroll
  for (int nt = 0; nt < 4; ++nt) {
    v8f acc = {};
    acc = wmma_bf16(a0, load_b_frag(wpack, 2, 0, nt, lane), acc);
    acc = wmma_bf16(a1, load_b_frag(wpack, 2, 1, nt, lane), acc);
    int c = nt * 16 + l15;
    float bi = bias[c];
    float partial = 0.f;
#pragma unroll
    for (int r = 0; r < 8; ++r) {
      size_t idx = ((size_t)(b * NNODE + row0 + r)) * DDIM + c;
      float m = hin[idx] + fmaxf(acc[r] + bi, 0.f);
      partial = fmaf(mk[r], m, partial);
    }
    pbuf[(((size_t)(b * MTILES + mt)) * 2 + half) * DDIM + c] = partial;
  }
}

// s[b][d] = sum over mt,half of partials  (deterministic fixed-order sum)
__global__ void nmp_reduce_s(const float* __restrict__ pbuf, float* __restrict__ s) {
  int t = blockIdx.x * blockDim.x + threadIdx.x;   // b*64 + d
  if (t >= BATCH * DDIM) return;
  int b = t >> 6, d = t & 63;
  float acc = 0.f;
  for (int mt = 0; mt < MTILES; ++mt)
    for (int hh = 0; hh < 2; ++hh)
      acc += pbuf[(((size_t)(b * MTILES + mt)) * 2 + hh) * DDIM + d];
  s[t] = acc;
}

// ---------------------------------------------------------------------------
// GRU cell over one (b, mt) tile per wave. h_new built from s and mask
// (h_new[i,:] = relu(mask[i]*s)). 48 WMMA per wave.
// ---------------------------------------------------------------------------
__global__ __launch_bounds__(128) void nmp_gru(
    const float* __restrict__ hin, const __bf16* __restrict__ wih,
    const __bf16* __restrict__ whh, const float* __restrict__ bih,
    const float* __restrict__ bhh, const float* __restrict__ sbuf,
    const float* __restrict__ mask, float* __restrict__ hout) {
  int wave = threadIdx.x >> 5, lane = threadIdx.x & 31;
  int tile = blockIdx.x * 4 + wave;
  int b = tile >> 5, mt = tile & 31;
  int half = lane >> 4, l15 = lane & 15;
  const float* hrow = hin + ((size_t)(b * NNODE + mt * 16 + l15)) * DDIM;
  v16bf ah0 = make_a_frag(hrow, 0, half);
  v16bf ah1 = make_a_frag(hrow, 1, half);
  // h_new A-fragment (row = l15): relu(mask[row] * s[K]), A-layout K mapping.
  float mrow = mask[b * NNODE + mt * 16 + l15];
  const float* sb = sbuf + b * DDIM;
  v16bf an0, an1;
#pragma unroll
  for (int e = 0; e < 16; ++e) {
    int kk = 8 * half + (e < 8 ? e : e + 8);
    an0[e] = (__bf16)fmaxf(mrow * sb[kk], 0.f);
    an1[e] = (__bf16)fmaxf(mrow * sb[32 + kk], 0.f);
  }
  int row0 = mt * 16 + 8 * half;
#pragma unroll
  for (int nt = 0; nt < 4; ++nt) {
    v8f ir = {}, iz = {}, inn = {}, hr = {}, hz = {}, hn = {};
    ir  = wmma_bf16(an0, load_b_frag(wih, 2, 0, nt,     lane), ir);
    ir  = wmma_bf16(an1, load_b_frag(wih, 2, 1, nt,     lane), ir);
    iz  = wmma_bf16(an0, load_b_frag(wih, 2, 0, nt + 4, lane), iz);
    iz  = wmma_bf16(an1, load_b_frag(wih, 2, 1, nt + 4, lane), iz);
    inn = wmma_bf16(an0, load_b_frag(wih, 2, 0, nt + 8, lane), inn);
    inn = wmma_bf16(an1, load_b_frag(wih, 2, 1, nt + 8, lane), inn);
    hr  = wmma_bf16(ah0, load_b_frag(whh, 2, 0, nt,     lane), hr);
    hr  = wmma_bf16(ah1, load_b_frag(whh, 2, 1, nt,     lane), hr);
    hz  = wmma_bf16(ah0, load_b_frag(whh, 2, 0, nt + 4, lane), hz);
    hz  = wmma_bf16(ah1, load_b_frag(whh, 2, 1, nt + 4, lane), hz);
    hn  = wmma_bf16(ah0, load_b_frag(whh, 2, 0, nt + 8, lane), hn);
    hn  = wmma_bf16(ah1, load_b_frag(whh, 2, 1, nt + 8, lane), hn);
    int c = nt * 16 + l15;
    float bir = bih[c], biz = bih[DDIM + c], bin = bih[2 * DDIM + c];
    float bhr = bhh[c], bhz = bhh[DDIM + c], bhn = bhh[2 * DDIM + c];
#pragma unroll
    for (int r = 0; r < 8; ++r) {
      size_t idx = ((size_t)(b * NNODE + row0 + r)) * DDIM + c;
      float hprev = hin[idx];
      float rg = 1.f / (1.f + __expf(-((ir[r] + bir) + (hr[r] + bhr))));
      float zg = 1.f / (1.f + __expf(-((iz[r] + biz) + (hz[r] + bhz))));
      float ng = tanhf((inn[r] + bin) + rg * (hn[r] + bhn));
      hout[idx] = ng + zg * (hprev - ng);
    }
  }
}

// ---------------------------------------------------------------------------
// Readout: hr = h + relu(h @ Wro + bro); v = relu(hr . fcw + fcb) per node.
// hr tile staged through LDS for the per-row dot product.
// ---------------------------------------------------------------------------
__global__ __launch_bounds__(128) void nmp_readout(
    const float* __restrict__ hin, const __bf16* __restrict__ wpack,
    const float* __restrict__ bias, const float* __restrict__ fcw,
    const float* __restrict__ fcb, float* __restrict__ vbuf) {
  __shared__ float tile_s[4][16 * DDIM];
  int wave = threadIdx.x >> 5, lane = threadIdx.x & 31;
  int tile = blockIdx.x * 4 + wave;
  int b = tile >> 5, mt = tile & 31;
  int half = lane >> 4, l15 = lane & 15;
  const float* hrow = hin + ((size_t)(b * NNODE + mt * 16 + l15)) * DDIM;
  v16bf a0 = make_a_frag(hrow, 0, half);
  v16bf a1 = make_a_frag(hrow, 1, half);
  int row0 = mt * 16 + 8 * half;
#pragma unroll
  for (int nt = 0; nt < 4; ++nt) {
    v8f acc = {};
    acc = wmma_bf16(a0, load_b_frag(wpack, 2, 0, nt, lane), acc);
    acc = wmma_bf16(a1, load_b_frag(wpack, 2, 1, nt, lane), acc);
    int c = nt * 16 + l15;
    float bi = bias[c];
#pragma unroll
    for (int r = 0; r < 8; ++r) {
      size_t idx = ((size_t)(b * NNODE + row0 + r)) * DDIM + c;
      tile_s[wave][(8 * half + r) * DDIM + c] = hin[idx] + fmaxf(acc[r] + bi, 0.f);
    }
  }
  __syncthreads();
  if (lane < 16) {
    float d = 0.f;
#pragma unroll
    for (int c = 0; c < DDIM; ++c) d = fmaf(tile_s[wave][lane * DDIM + c], fcw[c], d);
    vbuf[b * NNODE + mt * 16 + lane] = fmaxf(d + fcb[0], 0.f);
  }
}

// out[b] = mean over nodes of v
__global__ void nmp_mean(const float* __restrict__ vbuf, float* __restrict__ out) {
  int b = threadIdx.x;
  if (b >= BATCH) return;
  float acc = 0.f;
  for (int n = 0; n < NNODE; ++n) acc += vbuf[b * NNODE + n];
  out[b] = acc * (1.f / (float)NNODE);
}

// ---------------------------------------------------------------------------
extern "C" void kernel_launch(void* const* d_in, const int* in_sizes, int n_in,
                              void* d_out, int out_size, void* d_ws, size_t ws_size,
                              hipStream_t stream) {
  (void)in_sizes; (void)n_in; (void)out_size; (void)ws_size;
  const float* x        = (const float*)d_in[0];
  const float* mask     = (const float*)d_in[1];
  const float* emb_w    = (const float*)d_in[2];
  const float* emb_b    = (const float*)d_in[3];
  const float* res0_w   = (const float*)d_in[4];
  const float* res0_b   = (const float*)d_in[5];
  const float* msg_w    = (const float*)d_in[6];
  const float* msg_b    = (const float*)d_in[7];
  const float* gwih     = (const float*)d_in[8];
  const float* gwhh     = (const float*)d_in[9];
  const float* gbih     = (const float*)d_in[10];
  const float* gbhh     = (const float*)d_in[11];
  const float* ro_res_w = (const float*)d_in[12];
  const float* ro_res_b = (const float*)d_in[13];
  const float* ro_fc_w  = (const float*)d_in[14];
  const float* ro_fc_b  = (const float*)d_in[15];
  float* out = (float*)d_out;

  // Workspace layout
  char* ws = (char*)d_ws;
  const size_t HB = (size_t)BATCH * NNODE * DDIM * sizeof(float);   // 16 MB
  float*  hA    = (float*)ws;
  float*  hB    = (float*)(ws + HB);
  __bf16* wRes0 = (__bf16*)(ws + 2 * HB);
  __bf16* wRo   = wRes0 + 4096;                 // 8 frags * 512 elems
  __bf16* wMsg  = wRo + 4096;                   // 3 * 4096
  __bf16* wIh   = wMsg + 3 * 4096;              // 3 * 12288 (24 frags each)
  __bf16* wHh   = wIh + (size_t)3 * 12288;
  float*  pbuf  = (float*)(wHh + (size_t)3 * 12288);
  float*  sbuf  = pbuf + (size_t)BATCH * MTILES * 2 * DDIM;
  float*  vbuf  = sbuf + (size_t)BATCH * DDIM;

  // --- pack weights to bf16 B-fragments ---
  nmp_pack_b<<<8, 32, 0, stream>>>(res0_w,   wRes0, 2, 64, 64, 0);
  nmp_pack_b<<<8, 32, 0, stream>>>(ro_res_w, wRo,   2, 64, 64, 0);
  for (int i = 0; i < NITER; ++i) {
    nmp_pack_b<<<8,  32, 0, stream>>>(msg_w + (size_t)i * 64 * 64,  wMsg + (size_t)i * 4096,  2, 64, 64, 0);
    nmp_pack_b<<<24, 32, 0, stream>>>(gwih  + (size_t)i * 192 * 64, wIh  + (size_t)i * 12288, 2, 64, 192, 1);
    nmp_pack_b<<<24, 32, 0, stream>>>(gwhh  + (size_t)i * 192 * 64, wHh  + (size_t)i * 12288, 2, 64, 192, 1);
  }

  // --- embedding ---
  nmp_embed<<<(BATCH * NNODE * DDIM) / 256, 256, 0, stream>>>(x, emb_w, emb_b, hA);

  // --- residual block 0: hB = hA + relu(hA @ res0_w + b) ---
  nmp_resgemm<<<TILES_TOTAL / 4, 128, 0, stream>>>(hA, wRes0, res0_b, hB);

  float* cur = hB;
  float* other = hA;
  for (int i = 0; i < NITER; ++i) {
    nmp_msg<<<TILES_TOTAL / 4, 128, 0, stream>>>(
        cur, wMsg + (size_t)i * 4096, msg_b + (size_t)i * DDIM, mask, pbuf);
    nmp_reduce_s<<<(BATCH * DDIM) / 256, 256, 0, stream>>>(pbuf, sbuf);
    nmp_gru<<<TILES_TOTAL / 4, 128, 0, stream>>>(
        cur, wIh + (size_t)i * 12288, wHh + (size_t)i * 12288,
        gbih + (size_t)i * 3 * DDIM, gbhh + (size_t)i * 3 * DDIM,
        sbuf, mask, other);
    float* t = cur; cur = other; other = t;
  }

  // --- readout ---
  nmp_readout<<<TILES_TOTAL / 4, 128, 0, stream>>>(cur, wRo, ro_res_b, ro_fc_w, ro_fc_b, vbuf);
  nmp_mean<<<1, BATCH, 0, stream>>>(vbuf, out);
}
